// DotProductAttention_10900626997611
// MI455X (gfx1250) — matrix operand order repaired and checked by reading, not compiled, not run on hardware
//
#include <hip/hip_runtime.h>
#include <math.h>

typedef _Float16 h16 __attribute__((ext_vector_type(16)));
typedef _Float16 h2  __attribute__((ext_vector_type(2)));
typedef float    f8  __attribute__((ext_vector_type(8)));

#define QTILE 64   // query rows per workgroup (16 per wave, 4 waves)
#define KT    64   // keys per iteration (two WMMA K-steps for PV)
#define DH    64   // head dim

// ---- DPP16 cross-lane reduction within each 16-lane half (no DS traffic) ----
template<int CTRL>
__device__ __forceinline__ float dpp_mov(float x) {
    return __int_as_float(
        __builtin_amdgcn_update_dpp(0, __float_as_int(x), CTRL, 0xf, 0xf, true));
}
__device__ __forceinline__ float red16_max(float v) {
    v = fmaxf(v, dpp_mov<0xB1>(v));   // quad_perm [1,0,3,2]  : xor 1
    v = fmaxf(v, dpp_mov<0x4E>(v));   // quad_perm [2,3,0,1]  : xor 2
    v = fmaxf(v, dpp_mov<0x141>(v));  // row_half_mirror      : combine quads in 8
    v = fmaxf(v, dpp_mov<0x140>(v));  // row_mirror           : combine 8s in 16
    return v;
}
__device__ __forceinline__ float red16_sum(float v) {
    v += dpp_mov<0xB1>(v);
    v += dpp_mov<0x4E>(v);
    v += dpp_mov<0x141>(v);
    v += dpp_mov<0x140>(v);
    return v;
}

__global__ __launch_bounds__(128)
void fa_wmma_kernel(const float* __restrict__ Q,
                    const float* __restrict__ K,
                    const float* __restrict__ V,
                    const int*   __restrict__ VL,
                    float* __restrict__ O,
                    int S)
{
    // LDS: K tile [key][d], V tile transposed [d][key], per-wave P bounce [m][k]
    __shared__ __attribute__((aligned(16))) _Float16 ldsK[KT * DH];      // 8 KB
    __shared__ __attribute__((aligned(16))) _Float16 ldsV[DH * KT];      // 8 KB
    __shared__ __attribute__((aligned(16))) _Float16 ldsP[4 * 16 * KT];  // 8 KB

    const int tid  = threadIdx.x;
    const int wave = tid >> 5;
    const int lane = tid & 31;
    const int lo16 = lane & 15;
    const int hi   = lane >> 4;          // 0: lanes 0-15, 1: lanes 16-31

    const int b  = blockIdx.y;
    const int qb = blockIdx.x * QTILE + wave * 16;
    const int vl = VL[b];

    // fold 1/sqrt(64) and log2(e) into Q so softmax uses exp2 (v_exp_f32)
    const float qscale = 0.125f * 1.44269504088896340736f;

    // per-thread staging geometry: 8 float4s cover the 64x64 tile across 128 threads
    const int srow = tid >> 4;           // key row 0..7 base (row = srow + 8*i... see map)
    const int sd4  = (tid & 15) * 4;     // d 0..60 step 4
    const float* Kbase = K + ((size_t)b * S) * DH;
    const float* Vbase = V + ((size_t)b * S) * DH;

    // ---- Q A-fragments: 16x64 per wave -> two 16x32 f16 fragments ----
    // 16-bit A layout: lane holds row M=lane&15; VGPR-pair j covers K = kmap(j,hi)
    h16 aQ0, aQ1;
    {
        const float* qrow = Q + ((size_t)b * S + qb + lo16) * DH;
        #pragma unroll
        for (int j = 0; j < 8; ++j) {
            const int k0 = 2*j + ((j >= 4) ? 8 : 0) + (hi ? 8 : 0);
            const float2 x0 = *(const float2*)(qrow + k0);
            const float2 x1 = *(const float2*)(qrow + k0 + 32);
            aQ0[2*j]   = (_Float16)(x0.x * qscale);
            aQ0[2*j+1] = (_Float16)(x0.y * qscale);
            aQ1[2*j]   = (_Float16)(x1.x * qscale);
            aQ1[2*j+1] = (_Float16)(x1.y * qscale);
        }
    }

    f8 acc0 = {}, acc1 = {}, acc2 = {}, acc3 = {};
    float mrow[8], lrow[8];
    #pragma unroll
    for (int r = 0; r < 8; ++r) { mrow[r] = -INFINITY; lrow[r] = 0.0f; }

    // masked tail keys contribute exactly 0 after softmax -> skip whole tiles
    const int nTiles = (vl + KT - 1) / KT;   // uniform per block; S % KT == 0 so no OOB

    // ---- register prefetch of tile 0 (global_load_b128 into VGPRs) ----
    float4 kreg[8], vreg[8];
    #pragma unroll
    for (int i = 0; i < 8; ++i) {
        const size_t g = (size_t)(srow + 8 * i) * DH + sd4;
        kreg[i] = *(const float4*)(Kbase + g);
        vreg[i] = *(const float4*)(Vbase + g);
    }

    for (int t = 0; t < nTiles; ++t) {
        const int keyBase = t * KT;

        __syncthreads();   // all waves finished reading the previous tile's LDS
        // ---- convert prefetched regs -> LDS (K row-major f16, V transposed f16) ----
        #pragma unroll
        for (int i = 0; i < 8; ++i) {
            const int row = srow + 8 * i;            // key within tile 0..63
            h2 k01; k01[0] = (_Float16)kreg[i].x; k01[1] = (_Float16)kreg[i].y;
            h2 k23; k23[0] = (_Float16)kreg[i].z; k23[1] = (_Float16)kreg[i].w;
            *(h2*)&ldsK[row * DH + sd4]     = k01;
            *(h2*)&ldsK[row * DH + sd4 + 2] = k23;
            ldsV[(sd4 + 0) * KT + row] = (_Float16)vreg[i].x;
            ldsV[(sd4 + 1) * KT + row] = (_Float16)vreg[i].y;
            ldsV[(sd4 + 2) * KT + row] = (_Float16)vreg[i].z;
            ldsV[(sd4 + 3) * KT + row] = (_Float16)vreg[i].w;
        }
        // ---- issue next tile's global loads; they overlap this tile's compute ----
        if (t + 1 < nTiles) {
            const size_t tb = (size_t)(t + 1) * KT * DH;
            #pragma unroll
            for (int i = 0; i < 8; ++i) {
                const size_t g = tb + (size_t)(srow + 8 * i) * DH + sd4;
                kreg[i] = *(const float4*)(Kbase + g);
                vreg[i] = *(const float4*)(Vbase + g);
            }
        }
        __syncthreads();   // tile ready in LDS

        // ---- S = Q K^T : four 16x16 score tiles (keys sub*16 .. sub*16+15) ----
        // 16-bit B layout: lane holds column N=lane&15; VGPR-pair j covers K = 2j+(j>=4?8:0)+(hi?8:0)
        f8 s[4];
        #pragma unroll
        for (int sub = 0; sub < 4; ++sub) {
            const int keyLoc = sub * 16 + lo16;      // B column -> key row in ldsK
            h16 kb;
            f8 c = {};
            #pragma unroll
            for (int j = 0; j < 8; ++j) {
                const int kd = 2*j + ((j >= 4) ? 8 : 0) + (hi ? 8 : 0);
                const h2 p = *(const h2*)&ldsK[keyLoc * DH + kd];
                kb[2*j] = p[0]; kb[2*j+1] = p[1];
            }
            c = __builtin_amdgcn_wmma_f32_16x16x32_f16(false, aQ0, false, kb,
                                                       (short)0, c, false, false);
            #pragma unroll
            for (int j = 0; j < 8; ++j) {
                const int kd = 2*j + ((j >= 4) ? 8 : 0) + (hi ? 8 : 0) + 32;
                const h2 p = *(const h2*)&ldsK[keyLoc * DH + kd];
                kb[2*j] = p[0]; kb[2*j+1] = p[1];
            }
            c = __builtin_amdgcn_wmma_f32_16x16x32_f16(false, aQ1, false, kb,
                                                       (short)0, c, false, false);
            s[sub] = c;
        }

        // ---- mask invalid keys: only the (uniform) final partial tile pays this ----
        if (keyBase + KT > vl) {
            #pragma unroll
            for (int sub = 0; sub < 4; ++sub) {
                const bool msk = (keyBase + sub * 16 + lo16) >= vl;
                #pragma unroll
                for (int r = 0; r < 8; ++r)
                    s[sub][r] = msk ? -1.0e30f : s[sub][r];
            }
        }

        // ---- online softmax: row stats via DPP16 reductions ----
        float mnew[8], sc[8];
        #pragma unroll
        for (int r = 0; r < 8; ++r) {
            float v = fmaxf(fmaxf(s[0][r], s[1][r]), fmaxf(s[2][r], s[3][r]));
            v = red16_max(v);
            mnew[r] = fmaxf(mrow[r], v);
            sc[r]   = __builtin_amdgcn_exp2f(mrow[r] - mnew[r]);
            mrow[r] = mnew[r];
        }
        #pragma unroll
        for (int r = 0; r < 8; ++r) {
            float rs = 0.0f;
            #pragma unroll
            for (int sub = 0; sub < 4; ++sub) {
                const float p = __builtin_amdgcn_exp2f(s[sub][r] - mnew[r]);
                s[sub][r] = p;
                rs += p;
            }
            rs = red16_sum(rs);
            lrow[r] = lrow[r] * sc[r] + rs;
            acc0[r] *= sc[r]; acc1[r] *= sc[r];
            acc2[r] *= sc[r]; acc3[r] *= sc[r];
        }

        // ---- transpose P (C-layout) -> A-layout via per-wave LDS bounce ----
        _Float16* pbuf = &ldsP[wave * 16 * KT];
        #pragma unroll
        for (int r = 0; r < 8; ++r) {
            const int m = hi * 8 + r;
            #pragma unroll
            for (int sub = 0; sub < 4; ++sub)
                pbuf[m * KT + sub * 16 + lo16] = (_Float16)s[sub][r];
        }
        // LDS ops from one wave are in-order; wait for the stores to land
        asm volatile("s_wait_dscnt 0" ::: "memory");
        h16 pA0, pA1;
        #pragma unroll
        for (int j = 0; j < 8; ++j) {
            const int k0 = 2*j + ((j >= 4) ? 8 : 0) + (hi ? 8 : 0);
            const h2 p0 = *(const h2*)&pbuf[lo16 * KT + k0];
            const h2 p1 = *(const h2*)&pbuf[lo16 * KT + k0 + 32];
            pA0[2*j] = p0[0]; pA0[2*j+1] = p0[1];
            pA1[2*j] = p1[0]; pA1[2*j+1] = p1[1];
        }

        // ---- O += P V : 4 d-blocks x 2 K-steps, V B-fragments from transposed LDS ----
        #pragma unroll
        for (int dblk = 0; dblk < 4; ++dblk) {
            h16 vb0, vb1;
            #pragma unroll
            for (int j = 0; j < 8; ++j) {
                const int kk = 2*j + ((j >= 4) ? 8 : 0) + (hi ? 8 : 0);
                const h2 p0 = *(const h2*)&ldsV[(dblk * 16 + lo16) * KT + kk];
                const h2 p1 = *(const h2*)&ldsV[(dblk * 16 + lo16) * KT + kk + 32];
                vb0[2*j] = p0[0]; vb0[2*j+1] = p0[1];
                vb1[2*j] = p1[0]; vb1[2*j+1] = p1[1];
            }
            f8 a = (dblk == 0) ? acc0 : (dblk == 1) ? acc1 : (dblk == 2) ? acc2 : acc3;
            a = __builtin_amdgcn_wmma_f32_16x16x32_f16(false, pA0, false, vb0,
                                                       (short)0, a, false, false);
            a = __builtin_amdgcn_wmma_f32_16x16x32_f16(false, pA1, false, vb1,
                                                       (short)0, a, false, false);
            if      (dblk == 0) acc0 = a;
            else if (dblk == 1) acc1 = a;
            else if (dblk == 2) acc2 = a;
            else                acc3 = a;
        }
    }

    // ---- epilogue: O = acc / l ; C-layout lane holds column d, rows hi*8+r ----
    float* orow = O + ((size_t)b * S + qb) * DH;
    #pragma unroll
    for (int r = 0; r < 8; ++r) {
        const float rl = 1.0f / lrow[r];
        const int m = hi * 8 + r;
        orow[m * DH +  0 + lo16] = acc0[r] * rl;
        orow[m * DH + 16 + lo16] = acc1[r] * rl;
        orow[m * DH + 32 + lo16] = acc2[r] * rl;
        orow[m * DH + 48 + lo16] = acc3[r] * rl;
    }
}

extern "C" void kernel_launch(void* const* d_in, const int* in_sizes, int n_in,
                              void* d_out, int out_size, void* d_ws, size_t ws_size,
                              hipStream_t stream) {
    const float* Q  = (const float*)d_in[0];
    const float* K  = (const float*)d_in[1];
    const float* V  = (const float*)d_in[2];
    const int*   VL = (const int*)d_in[3];
    float* O = (float*)d_out;

    const int B = in_sizes[3];                      // one length per batch row
    const int S = in_sizes[0] / (B * DH);           // 2048
    dim3 grid(S / QTILE, B);
    fa_wmma_kernel<<<grid, 128, 0, stream>>>(Q, K, V, VL, O, S);
    (void)n_in; (void)out_size; (void)d_ws; (void)ws_size;
}
